// SpanClassifier_17154099380478
// MI455X (gfx1250) — compile-verified, matrix operand order checked by reading, not run on hardware
//
#include <hip/hip_runtime.h>
#include <hip/hip_bf16.h>
#include <math.h>

#define BB 8
#define LL 1024
#define DD 768
#define RR 16
#define MSP 4
#define NREL 128   // rel-embedding rows padded 33 -> 128 (full N tile, no guards)
#define ROWP 40    // LDS row pitch in bf16 elems (80B): conflict-free b128 frags

typedef __attribute__((ext_vector_type(16))) __bf16 v16bf;
typedef __attribute__((ext_vector_type(8)))  __bf16 v8bf;
typedef __attribute__((ext_vector_type(8)))  float  v8f;

union Frag { v16bf v; v8bf h[2]; };

static __device__ __forceinline__ void wait_async0() {
#if __has_builtin(__builtin_amdgcn_s_wait_asynccnt)
    __builtin_amdgcn_s_wait_asynccnt(0);
#else
    asm volatile("s_wait_asynccnt 0x0" ::: "memory");
#endif
}

// ---------------------------------------------------------------------------
// Tiled bf16 WMMA GEMM with double-buffered async-to-LDS pipeline.
//   A  : [M,K] bf16 row-major (lda, batch stride sA)
//   dA2/splitK: for k-columns >= splitK the A source shifts by the uniform
//               element delta dA2 (= A2 - A - splitK, requires lda2 == lda,
//               batch==1). Pass dA2=0, splitK=0 when unused.
//   BT : [N,K] bf16 row-major = B transposed (ldbt, batch stride sB)
//   C  : fp32 (Cf) and/or bf16 (Cb), shared ldc / sC
// flags: 1 = ReLU, 2 = scores epilogue (rel gather + attention mask)
// PRECONDITIONS: M % 128 == 0, N % 128 == 0, K % 32 == 0, splitK % 32 == 0.
// Pipeline: one s_wait_asynccnt 0 + one barrier per k-step; tile t+1 DMA
// overlaps tile t fragment loads + WMMAs.
// ---------------------------------------------------------------------------
__global__ __launch_bounds__(256) void k_gemm(
    const __bf16* __restrict__ A, int lda, long long sA,
    long long dA2, int splitK,
    const __bf16* __restrict__ BT, int ldbt, long long sB,
    float* __restrict__ Cf, __bf16* __restrict__ Cb, int ldc, long long sC,
    int M, int N, int K,
    const float* __restrict__ bias, float alpha, int flags,
    const float* __restrict__ qrel,
    const int* __restrict__ srcm, const int* __restrict__ amk,
    const int* __restrict__ spw, int iter)
{
    __shared__ __align__(16) __bf16 As[2][128 * ROWP];   // As[buf][row][k]
    __shared__ __align__(16) __bf16 Bs[2][128 * ROWP];   // Bs[buf][col][k]

    const int bz = blockIdx.z;
    const int tM = blockIdx.y * 128, tN = blockIdx.x * 128;
    const int t = threadIdx.x;
    const int lane = t & 31, wave = t >> 5;
    const int wr = wave >> 1, wc = wave & 1;           // wave tile: 32 rows x 64 cols

    // staging coords: each thread owns a 1-row x 32-byte chunk of each tile
    const int rs = t >> 1, cs = (t & 1) * 16;
    const __bf16* Abase = A + (long long)bz * sA + (long long)(tM + rs) * lda + cs;
    const __bf16* Bbase = BT + (long long)bz * sB + (long long)(tN + rs) * ldbt + cs;

    auto issue = [&](int buf, int k0) {
        // uniform scalar select: s_cselect_b64, no divergence
        const __bf16* pa = Abase + (long long)k0 + ((k0 >= splitK) ? dA2 : 0ll);
        const __bf16* pb = Bbase + k0;
        unsigned la = (unsigned)(size_t)&As[buf][rs * ROWP + cs];
        unsigned lb = (unsigned)(size_t)&Bs[buf][rs * ROWP + cs];
        asm volatile(
            "global_load_async_to_lds_b128 %0, %2, off\n\t"
            "global_load_async_to_lds_b128 %0, %2, off offset:16\n\t"
            "global_load_async_to_lds_b128 %1, %3, off\n\t"
            "global_load_async_to_lds_b128 %1, %3, off offset:16"
            :: "v"(la), "v"(lb), "v"(pa), "v"(pb) : "memory");
    };

    const v8f vz = {};
    v8f acc[2][4];
#pragma unroll
    for (int i = 0; i < 2; ++i)
#pragma unroll
        for (int j = 0; j < 4; ++j) acc[i][j] = vz;

    issue(0, 0);
    const int nt = K >> 5;
    for (int tt = 0; tt < nt; ++tt) {
        const int cur = tt & 1;
        wait_async0();       // only current tile's 4 async ops are outstanding
        __syncthreads();     // all waves: writes done + prior reads of !cur done
        if (tt + 1 < nt) issue(1 - cur, (tt + 1) << 5);   // overlap with compute

        const __bf16* Ac = As[cur];
        const __bf16* Bc = Bs[cur];
        const int kb = (lane < 16) ? 0 : 8;  // 16-bit A layout: lanes 0-15 K{0..7,16..23}
        const int ln = lane & 15;
        Frag af[2], bfr[4];
#pragma unroll
        for (int mi = 0; mi < 2; ++mi) {
            int row = wr * 32 + mi * 16 + ln;
            af[mi].h[0] = *(const v8bf*)(&Ac[row * ROWP + kb]);
            af[mi].h[1] = *(const v8bf*)(&Ac[row * ROWP + kb + 16]);
        }
#pragma unroll
        for (int ni = 0; ni < 4; ++ni) {
            int col = wc * 64 + ni * 16 + ln;
            bfr[ni].h[0] = *(const v8bf*)(&Bc[col * ROWP + kb]);
            bfr[ni].h[1] = *(const v8bf*)(&Bc[col * ROWP + kb + 16]);
        }
#pragma unroll
        for (int mi = 0; mi < 2; ++mi)
#pragma unroll
            for (int ni = 0; ni < 4; ++ni)
                acc[mi][ni] = __builtin_amdgcn_wmma_f32_16x16x32_bf16(
                    false, af[mi].v, false, bfr[ni].v, (short)0, acc[mi][ni],
                    false, false);
    }

    // Epilogue. C layout: lanes 0-15 -> M=vgpr,N=lane; lanes 16-31 -> M=vgpr+8.
    const int mb = (lane >= 16) ? 8 : 0;
    const int ln = lane & 15;
    float bv[4]; int cok[4]; int gnv[4];
#pragma unroll
    for (int ni = 0; ni < 4; ++ni) {
        int gn = tN + wc * 64 + ni * 16 + ln;
        gnv[ni] = gn;
        bv[ni] = bias ? bias[gn] : 0.0f;
        cok[ni] = (flags & 2) ? (amk[bz * N + gn] != 0) : 1;
    }
#pragma unroll
    for (int mi = 0; mi < 2; ++mi) {
#pragma unroll
        for (int j = 0; j < 8; ++j) {
            int gm = tM + wr * 32 + mi * 16 + mb + j;
            bool rok = true;
            const float* qr = nullptr;
            if (flags & 2) {
                rok = (srcm[bz * M + gm] != 0) && (iter < spw[bz * M + gm]);
                qr = qrel + ((long long)bz * M + gm) * NREL;
            }
            long long crow = (long long)bz * sC + (long long)gm * ldc;
#pragma unroll
            for (int ni = 0; ni < 4; ++ni) {
                float v = (acc[mi][ni][j] + bv[ni]) * alpha;
                if (flags & 1) v = fmaxf(v, 0.0f);
                if (flags & 2) {
                    int rel = gnv[ni] - gm;
                    rel = rel < -RR ? -RR : (rel > RR ? RR : rel);
                    v += qr[rel + RR];
                    if (!(rok && cok[ni])) v = -1e30f;
                }
                if (Cf) Cf[crow + gnv[ni]] = v;
                if (Cb) Cb[crow + gnv[ni]] = (__bf16)v;
            }
        }
    }
}

// In-place row softmax over 1024 elems (row stride rs in d_out), bf16 copy to ob.
__global__ __launch_bounds__(256) void k_softmax(
    float* __restrict__ S, long long rs, __bf16* __restrict__ ob)
{
    const int r = blockIdx.x;
    const int t = threadIdx.x;
    float* row = S + (long long)r * rs;
    __shared__ float red[256];
    float v[4];
    float mx = -3.4e38f;
#pragma unroll
    for (int j = 0; j < 4; ++j) { v[j] = row[t + j * 256]; mx = fmaxf(mx, v[j]); }
    red[t] = mx; __syncthreads();
    for (int s = 128; s > 0; s >>= 1) {
        if (t < s) red[t] = fmaxf(red[t], red[t + s]);
        __syncthreads();
    }
    mx = red[0]; __syncthreads();
    float sm = 0.0f;
#pragma unroll
    for (int j = 0; j < 4; ++j) { v[j] = __expf(v[j] - mx); sm += v[j]; }
    red[t] = sm; __syncthreads();
    for (int s = 128; s > 0; s >>= 1) {
        if (t < s) red[t] += red[t + s];
        __syncthreads();
    }
    float inv = 1.0f / red[0];
#pragma unroll
    for (int j = 0; j < 4; ++j) {
        float w = v[j] * inv;
        row[t + j * 256] = w;
        ob[(long long)r * LL + t + j * 256] = (__bf16)w;
    }
}

__global__ void k_cast(const float* __restrict__ in, __bf16* __restrict__ out,
                       long long n) {
    long long i = (long long)blockIdx.x * blockDim.x + threadIdx.x;
    if (i < n) out[i] = (__bf16)in[i];
}

// Transpose-cast: in [P,Q] f32 (batched) -> out [Q,P] bf16 (batched)
__global__ void k_castT(const float* __restrict__ in, __bf16* __restrict__ out,
                        int P, int Q, long long sIn, long long sOut) {
    long long i = (long long)blockIdx.x * blockDim.x + threadIdx.x;
    if (i >= (long long)P * Q) return;
    int b = blockIdx.y;
    int p = (int)(i % P);        // out column
    int q = (int)(i / P);        // out row
    out[(long long)b * sOut + (long long)q * P + p] =
        (__bf16)in[(long long)b * sIn + (long long)p * Q + q];
}

// relB[r][d] = rel[r][d] for r<=2R else 0; padded to NREL rows (N x K layout)
__global__ void k_relpad(const float* __restrict__ rel, __bf16* __restrict__ out) {
    int i = blockIdx.x * blockDim.x + threadIdx.x;
    if (i >= NREL * DD) return;
    int r = i / DD;
    out[i] = (r <= 2 * RR) ? (__bf16)rel[i] : (__bf16)0.0f;
}

__global__ void k_cnt(const int* __restrict__ amk, int* __restrict__ cnt) {
    __shared__ int red[256];
    int b = blockIdx.x, t = threadIdx.x;
    int c = 0;
    for (int k = t; k < LL; k += 256) c += (amk[b * LL + k] != 0);
    red[t] = c; __syncthreads();
    for (int s = 128; s > 0; s >>= 1) {
        if (t < s) red[t] += red[t + s];
        __syncthreads();
    }
    if (t == 0) cnt[b] = red[0];
}

// attn_w0 = amask / max(rowsum,1), stored bf16 into both st/ed current buffers
__global__ void k_init_attn(const int* __restrict__ srcm,
                            const int* __restrict__ amk,
                            const int* __restrict__ cnt,
                            __bf16* __restrict__ st, __bf16* __restrict__ ed) {
    long long i = (long long)blockIdx.x * blockDim.x + threadIdx.x;
    if (i >= (long long)BB * LL * LL) return;
    int k = (int)(i % LL);
    long long r = i / LL;
    int q = (int)(r % LL);
    int b = (int)(r / LL);
    float v = 0.0f;
    if (srcm[b * LL + q] != 0 && amk[b * LL + k] != 0) {
        int c = cnt[b]; if (c < 1) c = 1;
        v = 1.0f / (float)c;
    }
    __bf16 h = (__bf16)v;
    st[i] = h; ed[i] = h;
}

__global__ void k_masks(const int* __restrict__ spw, float* __restrict__ out) {
    int i = blockIdx.x * blockDim.x + threadIdx.x;
    if (i >= BB * LL * MSP) return;
    int s = i & 3, r = i >> 2;
    out[i] = (s < spw[r]) ? 1.0f : 0.0f;
}

extern "C" void kernel_launch(void* const* d_in, const int* in_sizes, int n_in,
                              void* d_out, int out_size, void* d_ws,
                              size_t ws_size, hipStream_t stream) {
    const float* hid    = (const float*)d_in[0];
    const float* srch   = (const float*)d_in[1];
    const int*   spw    = (const int*)d_in[2];
    const int*   amk    = (const int*)d_in[3];
    const int*   srcm   = (const int*)d_in[4];
    const float* Wq_st  = (const float*)d_in[6];
    const float* bq_st  = (const float*)d_in[7];
    const float* Wk_st  = (const float*)d_in[8];
    const float* bk_st  = (const float*)d_in[9];
    const float* rel_st = (const float*)d_in[10];
    const float* Wq_ed  = (const float*)d_in[11];
    const float* bq_ed  = (const float*)d_in[12];
    const float* Wk_ed  = (const float*)d_in[13];
    const float* bk_ed  = (const float*)d_in[14];
    const float* rel_ed = (const float*)d_in[15];
    const float* W_sp   = (const float*)d_in[16];
    const float* b_sp   = (const float*)d_in[17];
    float* out = (float*)d_out;

    const long long BLD = (long long)BB * LL * DD;
    const long long BLL = (long long)BB * LL * LL;
    const long long OUT_SEC = (long long)BB * LL * MSP * LL;

    size_t off = 0;
    char* W = (char*)d_ws;
    auto alloc = [&](size_t bytes) {
        off = (off + 255) & ~(size_t)255;
        size_t o = off; off += bytes; return o;
    };
    __bf16* hidb   = (__bf16*)(W + alloc(BLD * 2));   // hid bf16 [B,L,D]
    __bf16* hidTb  = (__bf16*)(W + alloc(BLD * 2));   // hid^T    [B,D,L]
    __bf16* hid1b  = (__bf16*)(W + alloc(BLD * 2));   // hidden ping
    __bf16* hid1x  = (__bf16*)(W + alloc(BLD * 2));   // hidden pong
    __bf16* hid2b  = (__bf16*)(W + alloc(BLD * 2));
    __bf16* hid2x  = (__bf16*)(W + alloc(BLD * 2));
    __bf16* q1b    = (__bf16*)(W + alloc(BLD * 2));   // hidc, then q
    __bf16* q2b    = (__bf16*)(W + alloc(BLD * 2));
    __bf16* k1b    = (__bf16*)(W + alloc(BLD * 2));   // k (BT layout for scores)
    __bf16* k2b    = (__bf16*)(W + alloc(BLD * 2));
    __bf16* WqstT  = (__bf16*)(W + alloc((size_t)DD * DD * 2));
    __bf16* WkstT  = (__bf16*)(W + alloc((size_t)DD * DD * 2));
    __bf16* WqedT  = (__bf16*)(W + alloc((size_t)DD * DD * 2));
    __bf16* WkedT  = (__bf16*)(W + alloc((size_t)DD * DD * 2));
    __bf16* WspT   = (__bf16*)(W + alloc((size_t)2 * DD * DD * 2)); // [D, 2D]
    __bf16* relBst = (__bf16*)(W + alloc((size_t)NREL * DD * 2));
    __bf16* relBed = (__bf16*)(W + alloc((size_t)NREL * DD * 2));
    float*  qrel1  = (float*)(W + alloc((size_t)BB * LL * NREL * 4));
    float*  qrel2  = (float*)(W + alloc((size_t)BB * LL * NREL * 4));
    __bf16* stc    = (__bf16*)(W + alloc(BLL * 2));
    __bf16* edc    = (__bf16*)(W + alloc(BLL * 2));
    int*    cnt    = (int*)(W + alloc(BB * 4));

    const int T = 256;
    auto nb = [](long long n) { return (unsigned)((n + 255) / 256); };

    auto gemm = [&](const __bf16* A, int lda, long long sA,
                    long long dA2, int splitK,
                    const __bf16* BT, int ldbt, long long sB,
                    float* Cf, __bf16* Cb, int ldc, long long sC,
                    int M, int N, int K, const float* bias, float alpha,
                    int flags, const float* qrel, int iter, int batch) {
        dim3 g((unsigned)(N / 128), (unsigned)(M / 128), (unsigned)batch);
        k_gemm<<<g, 256, 0, stream>>>(A, lda, sA, dA2, splitK,
                                      BT, ldbt, sB, Cf, Cb, ldc, sC,
                                      M, N, K, bias, alpha, flags, qrel,
                                      srcm, amk, spw, iter);
    };

    // ---- setup --------------------------------------------------------------
    k_cast<<<nb(BLD), T, 0, stream>>>(hid, hidb, BLD);
    k_cast<<<nb(BLD), T, 0, stream>>>(srch, hid1b, BLD);
    k_cast<<<nb(BLD), T, 0, stream>>>(srch, hid2b, BLD);
    // B operands stored transposed (N x K):
    k_castT<<<dim3(nb((long long)DD * DD), 1), T, 0, stream>>>(Wq_st, WqstT, DD, DD, 0, 0);
    k_castT<<<dim3(nb((long long)DD * DD), 1), T, 0, stream>>>(Wk_st, WkstT, DD, DD, 0, 0);
    k_castT<<<dim3(nb((long long)DD * DD), 1), T, 0, stream>>>(Wq_ed, WqedT, DD, DD, 0, 0);
    k_castT<<<dim3(nb((long long)DD * DD), 1), T, 0, stream>>>(Wk_ed, WkedT, DD, DD, 0, 0);
    k_castT<<<dim3(nb((long long)2 * DD * DD), 1), T, 0, stream>>>(W_sp, WspT, 2 * DD, DD, 0, 0);
    k_castT<<<dim3(nb((long long)LL * DD), BB), T, 0, stream>>>(
        hid, hidTb, LL, DD, (long long)LL * DD, (long long)DD * LL);
    k_relpad<<<nb((long long)NREL * DD), T, 0, stream>>>(rel_st, relBst);
    k_relpad<<<nb((long long)NREL * DD), T, 0, stream>>>(rel_ed, relBed);
    k_cnt<<<BB, 256, 0, stream>>>(amk, cnt);
    k_init_attn<<<nb(BLL), T, 0, stream>>>(srcm, amk, cnt, stc, edc);
    k_masks<<<nb((long long)BB * LL * MSP), T, 0, stream>>>(spw, out + 2 * OUT_SEC);

    // k = hid @ Wk + bk (loop-invariant); stored [B,L,D] == BT layout for scores
    gemm(hidb, DD, 0, 0, 0, WkstT, DD, 0, nullptr, k1b, DD, 0,
         BB * LL, DD, DD, bk_st, 1.0f, 0, nullptr, 0, 1);
    gemm(hidb, DD, 0, 0, 0, WkedT, DD, 0, nullptr, k2b, DD, 0,
         BB * LL, DD, DD, bk_ed, 1.0f, 0, nullptr, 0, 1);

    const float qscale = 1.0f / sqrtf((float)DD);
    __bf16 *h1c = hid1b, *h1n = hid1x;   // hidden ping-pong
    __bf16 *h2c = hid2b, *h2n = hid2x;

    // ---- span loop ----------------------------------------------------------
    for (int i = 0; i < MSP; ++i) {
        // hidc = attn @ hid  (batched; BT = hid^T [D,L]) -> q1b/q2b (temp)
        gemm(stc, LL, (long long)LL * LL, 0, 0,
             hidTb, LL, (long long)DD * LL,
             nullptr, q1b, DD, (long long)LL * DD,
             LL, DD, LL, nullptr, 1.0f, 0, nullptr, 0, BB);
        gemm(edc, LL, (long long)LL * LL, 0, 0,
             hidTb, LL, (long long)DD * LL,
             nullptr, q2b, DD, (long long)LL * DD,
             LL, DD, LL, nullptr, 1.0f, 0, nullptr, 0, BB);

        // hid' = relu(concat(hidp, hidc) @ W_sp + b_sp)  (split-A, no copy)
        gemm(h1c, DD, 0, (long long)(q1b - h1c) - DD, DD, WspT, 2 * DD, 0,
             nullptr, h1n, DD, 0,
             BB * LL, DD, 2 * DD, b_sp, 1.0f, 1, nullptr, 0, 1);
        gemm(h2c, DD, 0, (long long)(q2b - h2c) - DD, DD, WspT, 2 * DD, 0,
             nullptr, h2n, DD, 0,
             BB * LL, DD, 2 * DD, b_sp, 1.0f, 1, nullptr, 0, 1);

        // q = (hid' @ Wq + bq) / sqrt(D)   (overwrites the consumed hidc temp)
        gemm(h1n, DD, 0, 0, 0, WqstT, DD, 0, nullptr, q1b, DD, 0,
             BB * LL, DD, DD, bq_st, qscale, 0, nullptr, 0, 1);
        gemm(h2n, DD, 0, 0, 0, WqedT, DD, 0, nullptr, q2b, DD, 0,
             BB * LL, DD, DD, bq_ed, qscale, 0, nullptr, 0, 1);

        // qrel[b,q,r] = q . rel_emb[r]   (N = 128 padded)
        gemm(q1b, DD, 0, 0, 0, relBst, DD, 0, qrel1, nullptr, NREL, 0,
             BB * LL, NREL, DD, nullptr, 1.0f, 0, nullptr, 0, 1);
        gemm(q2b, DD, 0, 0, 0, relBed, DD, 0, qrel2, nullptr, NREL, 0,
             BB * LL, NREL, DD, nullptr, 1.0f, 0, nullptr, 0, 1);

        // scores = q @ k^T + qrel-gather, masked; written straight into d_out
        gemm(q1b, DD, (long long)LL * DD, 0, 0,
             k1b, DD, (long long)LL * DD,
             out + (long long)i * LL, nullptr, MSP * LL, (long long)LL * MSP * LL,
             LL, LL, DD, nullptr, 1.0f, 2, qrel1, i, BB);
        gemm(q2b, DD, (long long)LL * DD, 0, 0,
             k2b, DD, (long long)LL * DD,
             out + OUT_SEC + (long long)i * LL, nullptr, MSP * LL,
             (long long)LL * MSP * LL, LL, LL, DD, nullptr, 1.0f, 2, qrel2, i, BB);

        // softmax in place (fp32 in d_out) + bf16 copy for next iteration
        k_softmax<<<BB * LL, 256, 0, stream>>>(out + (long long)i * LL,
                                               (long long)MSP * LL, stc);
        k_softmax<<<BB * LL, 256, 0, stream>>>(out + OUT_SEC + (long long)i * LL,
                                               (long long)MSP * LL, edc);

        // swap hidden ping-pong
        __bf16* tmp;
        tmp = h1c; h1c = h1n; h1n = tmp;
        tmp = h2c; h2c = h2n; h2n = tmp;
    }
    (void)in_sizes; (void)n_in; (void)out_size; (void)ws_size;
}
